// LSTMSeq2Seq_28621662061202
// MI455X (gfx1250) — compile-verified, matrix operand order checked by reading, not run on hardware
//
#include <hip/hip_runtime.h>

#define DEVI __device__ __forceinline__

typedef unsigned short ushort_t;
typedef __attribute__((ext_vector_type(16))) __bf16 v16bf;
typedef __attribute__((ext_vector_type(8)))  float  v8f;

// Problem constants
constexpr int B_ = 64, T_ = 512, F_ = 64, H_ = 512, FUT_ = 96;
constexpr int G4  = 4 * H_;    // 2048 gate rows
constexpr int NWG = 16;        // workgroups in a scan (each owns 32 h-columns)
constexpr int HB  = B_ * H_;   // elements in one h buffer

// ---------------------------------------------------------------- helpers ---
DEVI ushort_t f2bf(float f) {
  union { float f; unsigned u; } v; v.f = f;
  unsigned u = v.u;
  u += 0x7fffu + ((u >> 16) & 1u);          // round to nearest even
  return (ushort_t)(u >> 16);
}

DEVI float sigm(float x) { return 1.f / (1.f + __expf(-x)); }
DEVI float tanh_f(float x) {
  x = fminf(15.f, fmaxf(-15.f, x));
  float e = __expf(2.f * x);
  return (e - 1.f) / (e + 1.f);
}

// Load a 16x32 bf16 WMMA operand fragment.
// p points at (row = lane%16) base; lane<16 -> K base 0, lane>=16 -> K base 8,
// VGPR4-7 hold K base+16 (ISA 7.12.2, 16-bit A-matrix 16x32).
DEVI v16bf frag16(const ushort_t* p) {
  union { uint4 q[2]; v16bf v; } u;
  u.q[0] = *(const uint4*)p;
  u.q[1] = *(const uint4*)(p + 16);
  return u.v;
}

DEVI v8f wmma_bf(v16bf a, v16bf b, v8f c) {
  return __builtin_amdgcn_wmma_f32_16x16x32_bf16(
      /*neg_a=*/false, a, /*neg_b=*/false, b,
      /*c_mod=*/(short)0, c, /*reuse_a=*/false, /*reuse_b=*/false);
}

// Async global -> LDS 16B copy (CDNA5 GLOBAL_LOAD_ASYNC_TO_LDS_B128, ASYNCcnt).
// lds_off = per-lane LDS byte address (low 32 bits of generic shared pointer,
// per the shared-aperture mapping).
DEVI void async_g2l_b128(unsigned lds_off, const void* gptr) {
  asm volatile("global_load_async_to_lds_b128 %0, %1, off"
               :: "v"(lds_off), "v"((unsigned long long)(size_t)gptr)
               : "memory");
}
DEVI void wait_asynccnt0() {
  asm volatile("s_wait_asynccnt 0x0" ::: "memory");
}

// Device-wide split barrier: monotone counter, acq/rel semantics.
DEVI void grid_barrier(unsigned* cnt, unsigned target) {
  __threadfence();
  __syncthreads();
  if (threadIdx.x == 0) {
    __hip_atomic_fetch_add(cnt, 1u, __ATOMIC_ACQ_REL, __HIP_MEMORY_SCOPE_AGENT);
    while (__hip_atomic_load(cnt, __ATOMIC_ACQUIRE, __HIP_MEMORY_SCOPE_AGENT) < target) {
      __builtin_amdgcn_s_sleep(8);
    }
  }
  __syncthreads();
}

// ---------------------------------------------------------------- prep ------
__global__ void k_f32_to_bf16(const float* __restrict__ s, ushort_t* __restrict__ d, int n) {
  int i = blockIdx.x * blockDim.x + threadIdx.x;
  if (i < n) d[i] = f2bf(s[i]);
}
__global__ void k_bias_add(const float* __restrict__ a, const float* __restrict__ b,
                           float* __restrict__ d, int n) {
  int i = blockIdx.x * blockDim.x + threadIdx.x;
  if (i < n) d[i] = a[i] + b[i];
}
__global__ void k_zero_u32(unsigned* __restrict__ p, int n) {
  int i = blockIdx.x * blockDim.x + threadIdx.x;
  if (i < n) p[i] = 0u;
}
__global__ void k_xdec_init(const float* __restrict__ seq, ushort_t* __restrict__ xdec) {
  int i = blockIdx.x * blockDim.x + threadIdx.x;
  if (i < B_ * F_) {
    int b = i >> 6, f = i & 63;
    xdec[i] = f2bf(seq[((size_t)b * T_ + (T_ - 1)) * F_ + f]);
  }
}

// ------------------------------------------------------- encoder LSTM scan --
// 16 WGs x 256 threads. WG w owns h-columns [32w, 32w+32) => 128 gate rows.
// Whh slice (128x512 bf16 = 128KB) is LDS-resident for all 512 steps, staged
// via async global->LDS DMA. Wave (m,cg): m=wave/2 -> batch rows 16m..16m+15;
// cg=wave&1 -> 16 h-columns. Each wave holds the i,f,g,o accumulator tiles for
// ITS columns -> elementwise LSTM update is pure-register; c persists in LDS.
__global__ __launch_bounds__(256) void lstm_scan_kernel(
    const ushort_t* __restrict__ xsrc, int x_rowstride, int ins,
    const ushort_t* __restrict__ wih,   // (2048, ins) bf16 row-major
    const ushort_t* __restrict__ whh,   // (2048, 512) bf16 row-major
    const float* __restrict__ bias,     // (2048) combined bih+bhh
    ushort_t* __restrict__ hbuf,        // 2*HB ping-pong, zero-initialized
    float* __restrict__ cfin,           // (64,512) final cell state out
    ushort_t* __restrict__ y_out,       // (B,T,H) bf16 or nullptr
    int steps, unsigned* __restrict__ sync_cnt)
{
  __shared__ __align__(16) ushort_t whhs[128 * 512];  // 128 KB
  __shared__ float cst[64 * 32];                      // 8 KB cell state

  const int hc0  = blockIdx.x * 32;
  const int tid  = threadIdx.x;
  const int wave = tid >> 5, lane = tid & 31;
  const int m = wave >> 1, cg = wave & 1;
  const int r = lane & 15, half = lane >> 4;

  // Stage Whh gate-slice into LDS with async DMA: LDS row (gate*32 + within)
  // <- global row (gate*512 + hc0 + within), 512 bf16 per row.
  for (int i = tid; i < 128 * 64; i += 256) {     // 64 uint4 chunks per row
    int row = i >> 6, chunk = i & 63;
    int gate = row >> 5, within = row & 31;
    const uint4* src = (const uint4*)(whh + (size_t)(gate * H_ + hc0 + within) * H_) + chunk;
    unsigned lds_off = (unsigned)(size_t)((const uint4*)whhs + row * 64 + chunk);
    async_g2l_b128(lds_off, src);
  }
  for (int i = tid; i < 64 * 32; i += 256) cst[i] = 0.f;
  wait_asynccnt0();
  __syncthreads();

  const v8f VZ = {0, 0, 0, 0, 0, 0, 0, 0};
  const int bcol = cg * 16 + r;     // column within this WG's 32-col slice
  const int gc   = hc0 + bcol;      // global h column
  const int lrow = cg * 16 + r;     // within-gate LDS row offset for B frags
  const bool wy = (y_out != nullptr);
  // Loop-invariant: combined biases for this lane's gate column.
  const float bi = bias[gc], bf = bias[H_ + gc],
              bg = bias[2 * H_ + gc], bo = bias[3 * H_ + gc];
  unsigned tgt = 0;

  for (int t = 0; t < steps; ++t) {
    const int rp = t & 1;
    const ushort_t* hread  = hbuf + rp * HB;
    ushort_t*       hwrite = hbuf + (rp ^ 1) * HB;

    v8f a0 = VZ, a1 = VZ, a2 = VZ, a3 = VZ;   // i,f,g,o tiles

    // x contribution (Wih streamed from L2)
    const ushort_t* xb = xsrc + (size_t)t * ins;
    const int wrow = hc0 + cg * 16 + r;       // weight row within each gate
    for (int kb = 0; kb < ins; kb += 32) {
      v16bf a = frag16(xb + (size_t)(m * 16 + r) * x_rowstride + kb + half * 8);
      a0 = wmma_bf(a, frag16(wih + (size_t)(0 * H_ + wrow) * ins + kb + half * 8), a0);
      a1 = wmma_bf(a, frag16(wih + (size_t)(1 * H_ + wrow) * ins + kb + half * 8), a1);
      a2 = wmma_bf(a, frag16(wih + (size_t)(2 * H_ + wrow) * ins + kb + half * 8), a2);
      a3 = wmma_bf(a, frag16(wih + (size_t)(3 * H_ + wrow) * ins + kb + half * 8), a3);
    }
    // h contribution (Whh in LDS)
    for (int kb = 0; kb < H_; kb += 32) {
      v16bf a = frag16(hread + (size_t)(m * 16 + r) * H_ + kb + half * 8);
      a0 = wmma_bf(a, frag16(whhs + (0 * 32 + lrow) * H_ + kb + half * 8), a0);
      a1 = wmma_bf(a, frag16(whhs + (1 * 32 + lrow) * H_ + kb + half * 8), a1);
      a2 = wmma_bf(a, frag16(whhs + (2 * 32 + lrow) * H_ + kb + half * 8), a2);
      a3 = wmma_bf(a, frag16(whhs + (3 * 32 + lrow) * H_ + kb + half * 8), a3);
    }

    #pragma unroll
    for (int j = 0; j < 8; ++j) {
      int brow = m * 16 + half * 8 + j;       // C/D layout: M = j + 8*(lane>=16)
      float iv = sigm(a0[j] + bi);
      float fv = sigm(a1[j] + bf);
      float gv = tanh_f(a2[j] + bg);
      float ov = sigm(a3[j] + bo);
      float c  = fv * cst[brow * 32 + bcol] + iv * gv;
      float h  = ov * tanh_f(c);
      cst[brow * 32 + bcol] = c;
      ushort_t hb = f2bf(h);
      hwrite[(size_t)brow * H_ + gc] = hb;
      if (wy) y_out[((size_t)brow * T_ + t) * H_ + gc] = hb;
    }

    tgt += NWG;
    grid_barrier(sync_cnt, tgt);
  }

  for (int i = tid; i < 64 * 32; i += 256) {
    int brow = i >> 5, cc = i & 31;
    cfin[(size_t)brow * H_ + hc0 + cc] = cst[i];
  }
}

// ----------------------------------------------------------- fused decoder --
// 96 steps: layer0 -> barrier -> layer1 -> barrier -> FC(+feedback) -> barrier.
// Weights stream from L2 (only 96 reuses); c-state for both layers in LDS.
__global__ __launch_bounds__(256) void lstm_decoder_kernel(
    ushort_t* __restrict__ xdec,                              // (64,64) bf16 feedback
    const ushort_t* __restrict__ wih0, const ushort_t* __restrict__ whh0,
    const float* __restrict__ b0,
    const ushort_t* __restrict__ wih1, const ushort_t* __restrict__ whh1,
    const float* __restrict__ b1,
    const ushort_t* __restrict__ fcw,  const float* __restrict__ fcb,
    ushort_t* __restrict__ h0buf, ushort_t* __restrict__ h1buf, // 2*HB each
    const float* __restrict__ cfin0, const float* __restrict__ cfin1,
    float* __restrict__ out, unsigned* __restrict__ sync_cnt)
{
  __shared__ float cst0[64 * 32];
  __shared__ float cst1[64 * 32];

  const int hc0  = blockIdx.x * 32;
  const int tid  = threadIdx.x;
  const int wave = tid >> 5, lane = tid & 31;
  const int m = wave >> 1, cg = wave & 1;
  const int r = lane & 15, half = lane >> 4;
  const int bcol = cg * 16 + r;
  const int gc   = hc0 + bcol;
  const int wrow = gc;

  for (int i = tid; i < 64 * 32; i += 256) {
    int brow = i >> 5, cc = i & 31;
    cst0[i] = cfin0[(size_t)brow * H_ + hc0 + cc];
    cst1[i] = cfin1[(size_t)brow * H_ + hc0 + cc];
  }
  __syncthreads();

  const v8f VZ = {0, 0, 0, 0, 0, 0, 0, 0};
  // Loop-invariant biases.
  const float bi0 = b0[gc], bf0 = b0[H_ + gc], bg0 = b0[2 * H_ + gc], bo0 = b0[3 * H_ + gc];
  const float bi1 = b1[gc], bf1 = b1[H_ + gc], bg1 = b1[2 * H_ + gc], bo1 = b1[3 * H_ + gc];
  const bool fc_active = (blockIdx.x < 4) && (wave < 4);   // wave-uniform
  const int fcol = blockIdx.x * 16 + r;
  const float bb = fc_active ? fcb[fcol] : 0.f;
  unsigned tgt = 0;

  for (int t = 0; t < FUT_; ++t) {
    const int rp = t & 1;

    // ---------------- layer 0: x = xdec (K=64) + h0 (K=512) ----------------
    {
      v8f a0 = VZ, a1 = VZ, a2 = VZ, a3 = VZ;
      for (int kb = 0; kb < F_; kb += 32) {
        v16bf a = frag16(xdec + (size_t)(m * 16 + r) * F_ + kb + half * 8);
        a0 = wmma_bf(a, frag16(wih0 + (size_t)(0 * H_ + wrow) * F_ + kb + half * 8), a0);
        a1 = wmma_bf(a, frag16(wih0 + (size_t)(1 * H_ + wrow) * F_ + kb + half * 8), a1);
        a2 = wmma_bf(a, frag16(wih0 + (size_t)(2 * H_ + wrow) * F_ + kb + half * 8), a2);
        a3 = wmma_bf(a, frag16(wih0 + (size_t)(3 * H_ + wrow) * F_ + kb + half * 8), a3);
      }
      const ushort_t* hread = h0buf + rp * HB;
      for (int kb = 0; kb < H_; kb += 32) {
        v16bf a = frag16(hread + (size_t)(m * 16 + r) * H_ + kb + half * 8);
        a0 = wmma_bf(a, frag16(whh0 + (size_t)(0 * H_ + wrow) * H_ + kb + half * 8), a0);
        a1 = wmma_bf(a, frag16(whh0 + (size_t)(1 * H_ + wrow) * H_ + kb + half * 8), a1);
        a2 = wmma_bf(a, frag16(whh0 + (size_t)(2 * H_ + wrow) * H_ + kb + half * 8), a2);
        a3 = wmma_bf(a, frag16(whh0 + (size_t)(3 * H_ + wrow) * H_ + kb + half * 8), a3);
      }
      ushort_t* hwrite = h0buf + (rp ^ 1) * HB;
      #pragma unroll
      for (int j = 0; j < 8; ++j) {
        int brow = m * 16 + half * 8 + j;
        float iv = sigm(a0[j] + bi0);
        float fv = sigm(a1[j] + bf0);
        float gv = tanh_f(a2[j] + bg0);
        float ov = sigm(a3[j] + bo0);
        float c  = fv * cst0[brow * 32 + bcol] + iv * gv;
        float h  = ov * tanh_f(c);
        cst0[brow * 32 + bcol] = c;
        hwrite[(size_t)brow * H_ + gc] = f2bf(h);
      }
    }
    tgt += NWG; grid_barrier(sync_cnt, tgt);

    // ---------------- layer 1: x = new h0 (K=512) + h1 (K=512) -------------
    {
      v8f a0 = VZ, a1 = VZ, a2 = VZ, a3 = VZ;
      const ushort_t* x1 = h0buf + (rp ^ 1) * HB;
      for (int kb = 0; kb < H_; kb += 32) {
        v16bf a = frag16(x1 + (size_t)(m * 16 + r) * H_ + kb + half * 8);
        a0 = wmma_bf(a, frag16(wih1 + (size_t)(0 * H_ + wrow) * H_ + kb + half * 8), a0);
        a1 = wmma_bf(a, frag16(wih1 + (size_t)(1 * H_ + wrow) * H_ + kb + half * 8), a1);
        a2 = wmma_bf(a, frag16(wih1 + (size_t)(2 * H_ + wrow) * H_ + kb + half * 8), a2);
        a3 = wmma_bf(a, frag16(wih1 + (size_t)(3 * H_ + wrow) * H_ + kb + half * 8), a3);
      }
      const ushort_t* hread = h1buf + rp * HB;
      for (int kb = 0; kb < H_; kb += 32) {
        v16bf a = frag16(hread + (size_t)(m * 16 + r) * H_ + kb + half * 8);
        a0 = wmma_bf(a, frag16(whh1 + (size_t)(0 * H_ + wrow) * H_ + kb + half * 8), a0);
        a1 = wmma_bf(a, frag16(whh1 + (size_t)(1 * H_ + wrow) * H_ + kb + half * 8), a1);
        a2 = wmma_bf(a, frag16(whh1 + (size_t)(2 * H_ + wrow) * H_ + kb + half * 8), a2);
        a3 = wmma_bf(a, frag16(whh1 + (size_t)(3 * H_ + wrow) * H_ + kb + half * 8), a3);
      }
      ushort_t* hwrite = h1buf + (rp ^ 1) * HB;
      #pragma unroll
      for (int j = 0; j < 8; ++j) {
        int brow = m * 16 + half * 8 + j;
        float iv = sigm(a0[j] + bi1);
        float fv = sigm(a1[j] + bf1);
        float gv = tanh_f(a2[j] + bg1);
        float ov = sigm(a3[j] + bo1);
        float c  = fv * cst1[brow * 32 + bcol] + iv * gv;
        float h  = ov * tanh_f(c);
        cst1[brow * 32 + bcol] = c;
        hwrite[(size_t)brow * H_ + gc] = f2bf(h);
      }
    }
    tgt += NWG; grid_barrier(sync_cnt, tgt);

    // ---------------- FC: out = h1_new @ fcW.T + fcb (64x64) ---------------
    if (fc_active) {                        // wave-uniform predicate, EXEC full
      v8f acc = VZ;
      const ushort_t* h1n = h1buf + (rp ^ 1) * HB;
      for (int kb = 0; kb < H_; kb += 32) {
        v16bf a = frag16(h1n + (size_t)(wave * 16 + r) * H_ + kb + half * 8);
        v16bf b = frag16(fcw + (size_t)fcol * H_ + kb + half * 8);
        acc = wmma_bf(a, b, acc);
      }
      #pragma unroll
      for (int j = 0; j < 8; ++j) {
        int brow = wave * 16 + half * 8 + j;
        float v = acc[j] + bb;
        out[((size_t)brow * FUT_ + t) * F_ + fcol] = v;      // (B,FUT,F) f32
        xdec[brow * F_ + fcol] = f2bf(v);                    // feedback
      }
    }
    tgt += NWG; grid_barrier(sync_cnt, tgt);
  }
}

// --------------------------------------------------------------- launcher ---
extern "C" void kernel_launch(void* const* d_in, const int* in_sizes, int n_in,
                              void* d_out, int out_size, void* d_ws, size_t ws_size,
                              hipStream_t stream) {
  (void)in_sizes; (void)n_in; (void)out_size; (void)ws_size;

  const float* input_seq = (const float*)d_in[0];
  const float* eWih0 = (const float*)d_in[1];
  const float* eWhh0 = (const float*)d_in[2];
  const float* ebih0 = (const float*)d_in[3];
  const float* ebhh0 = (const float*)d_in[4];
  const float* eWih1 = (const float*)d_in[5];
  const float* eWhh1 = (const float*)d_in[6];
  const float* ebih1 = (const float*)d_in[7];
  const float* ebhh1 = (const float*)d_in[8];
  const float* dWih0 = (const float*)d_in[9];
  const float* dWhh0 = (const float*)d_in[10];
  const float* dbih0 = (const float*)d_in[11];
  const float* dbhh0 = (const float*)d_in[12];
  const float* dWih1 = (const float*)d_in[13];
  const float* dWhh1 = (const float*)d_in[14];
  const float* dbih1 = (const float*)d_in[15];
  const float* dbhh1 = (const float*)d_in[16];
  const float* fcW   = (const float*)d_in[17];
  const float* fcb   = (const float*)d_in[18];

  char* ws = (char*)d_ws;
  size_t off = 0;
  auto alloc = [&](size_t bytes) -> void* {
    void* p = ws + off;
    off = (off + bytes + 255) & ~(size_t)255;
    return p;
  };

  unsigned* cnt   = (unsigned*)alloc(256);                 // 3 barrier counters
  ushort_t* h0buf = (ushort_t*)alloc((size_t)2 * HB * 2);  // ping-pong h, layer0
  ushort_t* h1buf = (ushort_t*)alloc((size_t)2 * HB * 2);  // ping-pong h, layer1
  const size_t zero_words = off / 4;                       // zero cnt + h bufs
  ushort_t* xseq  = (ushort_t*)alloc((size_t)B_ * T_ * F_ * 2);
  ushort_t* y0    = (ushort_t*)alloc((size_t)B_ * T_ * H_ * 2);
  ushort_t* wih0e = (ushort_t*)alloc((size_t)G4 * F_ * 2);
  ushort_t* whh0e = (ushort_t*)alloc((size_t)G4 * H_ * 2);
  ushort_t* wih1e = (ushort_t*)alloc((size_t)G4 * H_ * 2);
  ushort_t* whh1e = (ushort_t*)alloc((size_t)G4 * H_ * 2);
  ushort_t* wih0d = (ushort_t*)alloc((size_t)G4 * F_ * 2);
  ushort_t* whh0d = (ushort_t*)alloc((size_t)G4 * H_ * 2);
  ushort_t* wih1d = (ushort_t*)alloc((size_t)G4 * H_ * 2);
  ushort_t* whh1d = (ushort_t*)alloc((size_t)G4 * H_ * 2);
  ushort_t* fcwb  = (ushort_t*)alloc((size_t)F_ * H_ * 2);
  float*    be0   = (float*)alloc((size_t)G4 * 4);
  float*    be1   = (float*)alloc((size_t)G4 * 4);
  float*    bd0   = (float*)alloc((size_t)G4 * 4);
  float*    bd1   = (float*)alloc((size_t)G4 * 4);
  float*    cfin0 = (float*)alloc((size_t)HB * 4);
  float*    cfin1 = (float*)alloc((size_t)HB * 4);
  ushort_t* xdec  = (ushort_t*)alloc((size_t)B_ * F_ * 2);

  auto cvt = [&](const float* s, ushort_t* d, int n) {
    k_f32_to_bf16<<<(n + 255) / 256, 256, 0, stream>>>(s, d, n);
  };

  k_zero_u32<<<((int)zero_words + 255) / 256, 256, 0, stream>>>(cnt, (int)zero_words);
  cvt(input_seq, xseq, B_ * T_ * F_);
  cvt(eWih0, wih0e, G4 * F_);
  cvt(eWhh0, whh0e, G4 * H_);
  cvt(eWih1, wih1e, G4 * H_);
  cvt(eWhh1, whh1e, G4 * H_);
  cvt(dWih0, wih0d, G4 * F_);
  cvt(dWhh0, whh0d, G4 * H_);
  cvt(dWih1, wih1d, G4 * H_);
  cvt(dWhh1, whh1d, G4 * H_);
  cvt(fcW, fcwb, F_ * H_);
  k_bias_add<<<(G4 + 255) / 256, 256, 0, stream>>>(ebih0, ebhh0, be0, G4);
  k_bias_add<<<(G4 + 255) / 256, 256, 0, stream>>>(ebih1, ebhh1, be1, G4);
  k_bias_add<<<(G4 + 255) / 256, 256, 0, stream>>>(dbih0, dbhh0, bd0, G4);
  k_bias_add<<<(G4 + 255) / 256, 256, 0, stream>>>(dbih1, dbhh1, bd1, G4);
  k_xdec_init<<<(B_ * F_ + 255) / 256, 256, 0, stream>>>(input_seq, xdec);

  // Encoder layer 0: x = input_seq (B,T,F), writes y0 for layer 1.
  lstm_scan_kernel<<<NWG, 256, 0, stream>>>(
      xseq, T_ * F_, F_, wih0e, whh0e, be0, h0buf, cfin0, y0, T_, cnt + 0);
  // Encoder layer 1: x = y0 (B,T,H).
  lstm_scan_kernel<<<NWG, 256, 0, stream>>>(
      y0, T_ * H_, H_, wih1e, whh1e, be1, h1buf, cfin1, nullptr, T_, cnt + 1);
  // Decoder: consumes final h (left in h0buf[0]/h1buf[0]) and final c.
  lstm_decoder_kernel<<<NWG, 256, 0, stream>>>(
      xdec, wih0d, whh0d, bd0, wih1d, whh1d, bd1, fcwb, fcb,
      h0buf, h1buf, cfin0, cfin1, (float*)d_out, cnt + 2);
}
